// Net1_34351148433854
// MI455X (gfx1250) — matrix operand order repaired and checked by reading, not compiled, not run on hardware
//
#include <hip/hip_runtime.h>
#include <hip/hip_bf16.h>
#include <math.h>

// ---------------------------------------------------------------------------
// GAT double-backbone + FC head for MI455X (gfx1250, wave32, WMMA).
// Dense matmuls -> v_wmma_f32_16x16x32_f16 (weights pre-transposed so both
// LDS tile stages are contiguous float4 loads); edge softmax/scatter -> atomics.
// Param leaves assumed flattened in jax tree order (sorted dict keys at every
// level): top-level inputs in insertion order, `params` expanded to:
//   [6..11]  fbn1..3 {b,g}
//   [12..19] fc1..fc4 {W,b}
//   [20..63] g1: bn1..4{b,g}, conv1..4{W,We,a_dst,a_edge,a_src,bias},
//                rbn1..3{b,g}, res1..3{W,b}
//   [64..107] g2 (same layout)
//   [108..113] rbn1..3 {b,g}
//   [114..119] rfc1..3 {W,b}
// The 4 int arrays are taken from the tail: d_in[n_in-4..n_in-1].
// ---------------------------------------------------------------------------

typedef __attribute__((ext_vector_type(16))) _Float16 v16h;
typedef __attribute__((ext_vector_type(8)))  float    v8f;

#define HC 256
#define CH 64
#define GDIM 20
#define NG 512
#define CATD 552   // (256+20)*2
#define LDK 72     // LDS row pitch in halves (144B, 16B aligned, conflict-skewed)

__device__ __forceinline__ void st4h(_Float16* dst, float x, float y, float z, float w) {
  union { _Float16 h[4]; float2 f; } u;
  u.h[0] = (_Float16)x; u.h[1] = (_Float16)y;
  u.h[2] = (_Float16)z; u.h[3] = (_Float16)w;
  *(float2*)dst = u.f;               // single 8B ds_store
}

// ------------------------- weight transpose --------------------------------
// WT[N][K] = W[K][N]; tiny (weights <= 552x512), done once per GEMM call.
__global__ void __launch_bounds__(256)
k_transpose(const float* __restrict__ W, float* __restrict__ WT, int K, int N) {
  __shared__ float tile[32][33];
  const int k0 = blockIdx.x * 32, n0 = blockIdx.y * 32;
  const int tx = threadIdx.x & 31, ty = threadIdx.x >> 5;   // 32 x 8
  for (int i = ty; i < 32; i += 8) {
    int k = k0 + i, n = n0 + tx;
    tile[i][tx] = (k < K && n < N) ? W[(long)k * N + n] : 0.f;
  }
  __syncthreads();
  for (int i = ty; i < 32; i += 8) {
    int n = n0 + i, k = k0 + tx;
    if (n < N && k < K) WT[(long)n * K + k] = tile[tx][i];
  }
}

// ------------------------- WMMA GEMM ---------------------------------------
// C[M,N] = A[M,K] @ B[K,N] (+bias[n]), B given transposed (BT[N][K]).
// f32 in/out, f16 WMMA, f32 acc. Block 256thr = 8 wave32, tile 32(M)x64(N),
// K stepped by 64 -> 2 v_wmma per barrier interval.
__global__ void __launch_bounds__(256)
k_gemm(const float* __restrict__ A, const float* __restrict__ BT,
       const float* __restrict__ bias, float* __restrict__ C,
       int M, int N, int K)
{
  __shared__ __align__(16) _Float16 As[32 * LDK];
  __shared__ __align__(16) _Float16 Bs[64 * LDK];
  const int tid  = threadIdx.x;
  const int lane = tid & 31, wave = tid >> 5;
  const int half = lane >> 4, r = lane & 15;
  const int wm = wave >> 2, wn = wave & 3;          // wave grid 2 x 4
  const int gm0 = blockIdx.y * 32;
  const int gn0 = blockIdx.x * 64;

  v8f acc = {0.f, 0.f, 0.f, 0.f, 0.f, 0.f, 0.f, 0.f};

  for (int k0 = 0; k0 < K; k0 += 64) {
    const bool fullK = (k0 + 64 <= K);
    // ---- A tile: 32 rows x 64 k ---- (uniform fast/slow branch)
    if (fullK && (gm0 + 32 <= M)) {
#pragma unroll
      for (int i = 0; i < 2; ++i) {
        int idx = tid + i * 256, row = idx >> 4, k4 = (idx & 15) * 4;
        const float4 v = *(const float4*)&A[(long)(gm0 + row) * K + k0 + k4];
        st4h(&As[row * LDK + k4], v.x, v.y, v.z, v.w);
      }
    } else {
#pragma unroll
      for (int i = 0; i < 2; ++i) {
        int idx = tid + i * 256, row = idx >> 4, k4 = (idx & 15) * 4;
        int gr = gm0 + row;
        float v[4];
#pragma unroll
        for (int j = 0; j < 4; ++j) {
          int gk = k0 + k4 + j;
          v[j] = (gr < M && gk < K) ? A[(long)gr * K + gk] : 0.f;
        }
        st4h(&As[row * LDK + k4], v[0], v[1], v[2], v[3]);
      }
    }
    // ---- B tile: 64 n-rows x 64 k (from BT[N][K]) ----
    if (fullK && (gn0 + 64 <= N)) {
#pragma unroll
      for (int i = 0; i < 4; ++i) {
        int idx = tid + i * 256, row = idx >> 4, k4 = (idx & 15) * 4;
        const float4 v = *(const float4*)&BT[(long)(gn0 + row) * K + k0 + k4];
        st4h(&Bs[row * LDK + k4], v.x, v.y, v.z, v.w);
      }
    } else {
#pragma unroll
      for (int i = 0; i < 4; ++i) {
        int idx = tid + i * 256, row = idx >> 4, k4 = (idx & 15) * 4;
        int gn = gn0 + row;
        float v[4];
#pragma unroll
        for (int j = 0; j < 4; ++j) {
          int gk = k0 + k4 + j;
          v[j] = (gn < N && gk < K) ? BT[(long)gn * K + gk] : 0.f;
        }
        st4h(&Bs[row * LDK + k4], v[0], v[1], v[2], v[3]);
      }
    }
    __syncthreads();

    union { v16h v; uint4 q[2]; } fa, fb;
    const _Float16* arow = &As[(wm * 16 + r) * LDK];
    const _Float16* bcol = &Bs[(wn * 16 + r) * LDK];
#pragma unroll
    for (int kk = 0; kk < 2; ++kk) {
      const int o = kk * 32;
      // A frag: K = o + (e/8)*16 + half*8 + e%8
      fa.q[0] = *(const uint4*)(arow + o + half * 8);
      fa.q[1] = *(const uint4*)(arow + o + 16 + half * 8);
      // B frag: K = o + (e/8)*16 + half*8 + e%8
      fb.q[0] = *(const uint4*)(bcol + o + half * 8);
      fb.q[1] = *(const uint4*)(bcol + o + 16 + half * 8);
      acc = __builtin_amdgcn_wmma_f32_16x16x32_f16(false, fa.v, false, fb.v,
                                                   (short)0, acc, false, false);
    }
    __syncthreads();
  }
  // D layout: VGPR t -> M = half*8 + t, N = r
#pragma unroll
  for (int t = 0; t < 8; ++t) {
    int row = gm0 + wm * 16 + half * 8 + t;
    int col = gn0 + wn * 16 + r;
    if (row < M && col < N) {
      float o = acc[t];
      if (bias) o += bias[col];
      C[(long)row * N + col] = o;
    }
  }
}

// ------------------------- small kernels -----------------------------------
__global__ void k_fill(float* p, float v, long n) {
  long i = (long)blockIdx.x * blockDim.x + threadIdx.x;
  if (i < n) p[i] = v;
}

// Me[f,h] = sum_c We[f, h*64+c] * a_edge[h,c]
__global__ void k_fold_edge(const float* __restrict__ We,
                            const float* __restrict__ ae, float* __restrict__ Me) {
  int tid = threadIdx.x;
  if (tid >= 64) return;
  int f = tid >> 2, h = tid & 3;
  float s = 0.f;
  for (int c = 0; c < CH; ++c) s += We[f * HC + h * CH + c] * ae[h * CH + c];
  Me[f * 4 + h] = s;
}

// a_e[e,h] = ea[e,:] @ Me[:,h]
__global__ void k_edge_ae(const float* __restrict__ ea, const float* __restrict__ Me,
                          float* __restrict__ out, int E) {
  long i = (long)blockIdx.x * blockDim.x + threadIdx.x;
  if (i >= (long)E * 4) return;
  int e = (int)(i >> 2), h = (int)(i & 3);
  const float* row = &ea[(long)e * 16];
  float s = 0.f;
#pragma unroll
  for (int f = 0; f < 16; ++f) s += row[f] * Me[f * 4 + h];
  out[i] = s;
}

// a[n,h] = sum_c xw[n, h*64+c] * av[h*64+c]
__global__ void k_node_coef(const float* __restrict__ xw, const float* __restrict__ av,
                            float* __restrict__ out, int n) {
  long i = (long)blockIdx.x * blockDim.x + threadIdx.x;
  if (i >= (long)n * 4) return;
  int node = (int)(i >> 2), h = (int)(i & 3);
  const float* row = &xw[(long)node * HC + h * CH];
  const float* a = &av[h * CH];
  float s = 0.f;
#pragma unroll 8
  for (int c = 0; c < CH; ++c) s += row[c] * a[c];
  out[i] = s;
}

__device__ __forceinline__ void atomicMaxFloat(float* addr, float value) {
  if (value >= 0.f) atomicMax((int*)addr, __float_as_int(value));
  else              atomicMin((unsigned int*)addr, __float_as_uint(value));
}

__global__ void k_edge_alpha(const int* __restrict__ ei, const float* __restrict__ as_,
                             const float* __restrict__ ad_, const float* __restrict__ ae_,
                             float* __restrict__ alpha, float* __restrict__ mmax, int E) {
  long i = (long)blockIdx.x * blockDim.x + threadIdx.x;
  if (i >= (long)E * 4) return;
  int e = (int)(i >> 2), h = (int)(i & 3);
  int s = ei[e], d = ei[E + e];
  float a = as_[s * 4 + h] + ad_[d * 4 + h] + ae_[i];
  a = a > 0.f ? a : 0.2f * a;                       // leaky_relu(0.2)
  alpha[i] = a;
  atomicMaxFloat(&mmax[d * 4 + h], a);
}

__global__ void k_edge_expsum(const int* __restrict__ ei, const float* __restrict__ mmax,
                              float* __restrict__ alpha, float* __restrict__ ssum, int E) {
  long i = (long)blockIdx.x * blockDim.x + threadIdx.x;
  if (i >= (long)E * 4) return;
  int e = (int)(i >> 2), h = (int)(i & 3);
  int d = ei[E + e];
  float ex = __expf(alpha[i] - mmax[d * 4 + h]);
  alpha[i] = ex;                                    // alpha now holds exp()
  atomicAdd(&ssum[d * 4 + h], ex);
}

// out[dst] += xw[src] * (exp / sum); 64 lanes/edge, float4 per lane
__global__ void k_edge_scatter(const int* __restrict__ ei, const float* __restrict__ xw,
                               const float* __restrict__ wexp, const float* __restrict__ ssum,
                               float* __restrict__ out, int E) {
  long i = (long)blockIdx.x * blockDim.x + threadIdx.x;
  if (i >= (long)E * 64) return;
  int e = (int)(i >> 6), l = (int)(i & 63);
  int c = l * 4, h = l >> 4;
  int s = ei[e], d = ei[E + e];
  float w = wexp[e * 4 + h] / (ssum[d * 4 + h] + 1e-16f);
  const float4 v = *(const float4*)&xw[(long)s * HC + c];
  float* o = &out[(long)d * HC + c];
  atomicAdd(o + 0, v.x * w); atomicAdd(o + 1, v.y * w);
  atomicAdd(o + 2, v.z * w); atomicAdd(o + 3, v.w * w);
}

__global__ void k_bias_relu(float* __restrict__ x, const float* __restrict__ bias,
                            long total, int C) {
  long i = (long)blockIdx.x * blockDim.x + threadIdx.x;
  if (i >= total) return;
  int c = (int)(i % C);
  float v = x[i] + (bias ? bias[c] : 0.f);
  x[i] = v > 0.f ? v : 0.f;
}

// per-column sum / sumsq into st[0..C-1], st[C..2C-1] (pre-zeroed)
__global__ void k_colstats(const float* __restrict__ x, float* __restrict__ st,
                           int M, int C) {
  for (int col = threadIdx.x; col < C; col += blockDim.x) {
    float s = 0.f, s2 = 0.f;
    for (int r = blockIdx.x; r < M; r += gridDim.x) {
      float v = x[(long)r * C + col];
      s += v; s2 += v * v;
    }
    atomicAdd(&st[col], s);
    atomicAdd(&st[C + col], s2);
  }
}

// out = BN(t) [+ BN(r)]  (training-mode BN from raw sums)
__global__ void k_bn2_add(const float* __restrict__ t, const float* __restrict__ stt,
                          const float* __restrict__ gt, const float* __restrict__ bt,
                          const float* __restrict__ r, const float* __restrict__ str,
                          const float* __restrict__ gr, const float* __restrict__ br,
                          float* __restrict__ out, int M, int C) {
  long i = (long)blockIdx.x * blockDim.x + threadIdx.x;
  if (i >= (long)M * C) return;
  int c = (int)(i % C);
  float inv = 1.0f / (float)M;
  float mu = stt[c] * inv, var = stt[C + c] * inv - mu * mu;
  float y = (t[i] - mu) * rsqrtf(var + 1e-5f) * gt[c] + bt[c];
  if (r) {
    float mur = str[c] * inv, varr = str[C + c] * inv - mur * mur;
    y += (r[i] - mur) * rsqrtf(varr + 1e-5f) * gr[c] + br[c];
  }
  out[i] = y;
}

__global__ void k_pool(const float* __restrict__ h, const int* __restrict__ batch,
                       float* __restrict__ xcat, int n, int colbase) {
  long i = (long)blockIdx.x * blockDim.x + threadIdx.x;
  if (i >= (long)n * HC) return;
  int node = (int)(i >> 8), c = (int)(i & 255);
  atomicAdd(&xcat[(long)batch[node] * CATD + colbase + c], h[i]);
}

__global__ void k_copycols(const float* __restrict__ g, float* __restrict__ xcat,
                           int rows, int cols, int colbase) {
  int i = blockIdx.x * blockDim.x + threadIdx.x;
  if (i >= rows * cols) return;
  int r = i / cols, c = i % cols;
  xcat[(long)r * CATD + colbase + c] = g[i];
}

__global__ void k_fc4(const float* __restrict__ x, const float* __restrict__ W,
                      const float* __restrict__ b, float* __restrict__ out, int M, int K) {
  int r = blockIdx.x * blockDim.x + threadIdx.x;
  if (r >= M) return;
  float s = b[0];
  for (int k = 0; k < K; ++k) s += x[(long)r * K + k] * W[k];
  out[r] = s;
}

// ------------------------- host orchestration ------------------------------
static inline unsigned cdiv(long a, long b) { return (unsigned)((a + b - 1) / b); }

struct Scratch {
  float *h, *xw, *convo, *R, *wT;
  float *a_s, *a_d, *mmax, *ssum, *a_e, *alpha, *Me;
  float *xcat, *fA, *fB, *fR, *fT, *stT, *stR;
};

// transpose weight then WMMA GEMM
static void gemmT(const float* A, const float* W, const float* bias, float* C,
                  int M, int N, int K, const Scratch& w, hipStream_t st) {
  k_transpose<<<dim3(cdiv(K, 32), cdiv(N, 32)), 256, 0, st>>>(W, w.wT, K, N);
  k_gemm<<<dim3(N / 64, cdiv(M, 32)), 256, 0, st>>>(A, w.wT, bias, C, M, N, K);
}

static void run_backbone(void* const* d_in, int base, const float* x_in,
                         const float* ea, const int* ei, const int* batch,
                         int n, int E, int colbase, const Scratch& w,
                         hipStream_t st) {
  auto P = [&](int i) { return (const float*)d_in[i]; };
  const long nodes = (long)n * HC;
  const dim3 gNodes(cdiv(nodes, 256)), gE4(cdiv((long)E * 4, 256));

  for (int layer = 1; layer <= 4; ++layer) {
    const float* Ain = (layer == 1) ? x_in : w.h;
    int Kin = (layer == 1) ? 64 : HC;
    int cb = base + 8 + (layer - 1) * 6;        // conv leaves
    bool has_res = (layer <= 3);

    if (has_res) {                               // residual linear + its BN stats
      int rb = base + 38 + (layer - 1) * 2;      // res{W,b}
      gemmT(Ain, P(rb), P(rb + 1), w.R, n, HC, Kin, w, st);
      k_fill<<<dim3(2), 256, 0, st>>>(w.stR, 0.f, 2 * HC);
      k_colstats<<<dim3(256), 256, 0, st>>>(w.R, w.stR, n, HC);
    }
    // node transform (WMMA)
    gemmT(Ain, P(cb + 0), nullptr, w.xw, n, HC, Kin, w, st);
    // attention coefficients
    k_node_coef<<<cdiv((long)n * 4, 256), 256, 0, st>>>(w.xw, P(cb + 4), w.a_s, n);
    k_node_coef<<<cdiv((long)n * 4, 256), 256, 0, st>>>(w.xw, P(cb + 2), w.a_d, n);
    k_fold_edge<<<1, 64, 0, st>>>(P(cb + 1), P(cb + 3), w.Me);
    k_edge_ae<<<gE4, 256, 0, st>>>(ea, w.Me, w.a_e, E);
    // softmax over in-edges
    k_fill<<<cdiv((long)n * 4, 256), 256, 0, st>>>(w.mmax, -INFINITY, (long)n * 4);
    k_fill<<<cdiv((long)n * 4, 256), 256, 0, st>>>(w.ssum, 0.f, (long)n * 4);
    k_fill<<<gNodes, 256, 0, st>>>(w.convo, 0.f, nodes);
    k_edge_alpha<<<gE4, 256, 0, st>>>(ei, w.a_s, w.a_d, w.a_e, w.alpha, w.mmax, E);
    k_edge_expsum<<<gE4, 256, 0, st>>>(ei, w.mmax, w.alpha, w.ssum, E);
    k_edge_scatter<<<cdiv((long)E * 64, 256), 256, 0, st>>>(ei, w.xw, w.alpha,
                                                            w.ssum, w.convo, E);
    // bias + relu + BN (+ BN'd residual)
    k_bias_relu<<<gNodes, 256, 0, st>>>(w.convo, P(cb + 5), nodes, HC);
    k_fill<<<dim3(2), 256, 0, st>>>(w.stT, 0.f, 2 * HC);
    k_colstats<<<dim3(256), 256, 0, st>>>(w.convo, w.stT, n, HC);
    int bnb = base + (layer - 1) * 2;            // bn{b,g}
    if (has_res) {
      int rbn = base + 32 + (layer - 1) * 2;     // rbn{b,g}
      k_bn2_add<<<gNodes, 256, 0, st>>>(w.convo, w.stT, P(bnb + 1), P(bnb),
                                        w.R, w.stR, P(rbn + 1), P(rbn),
                                        w.h, n, HC);
    } else {
      k_bn2_add<<<gNodes, 256, 0, st>>>(w.convo, w.stT, P(bnb + 1), P(bnb),
                                        nullptr, nullptr, nullptr, nullptr,
                                        w.h, n, HC);
    }
  }
  k_pool<<<gNodes, 256, 0, st>>>(w.h, batch, w.xcat, n, colbase);
}

extern "C" void kernel_launch(void* const* d_in, const int* in_sizes, int n_in,
                              void* d_out, int out_size, void* d_ws, size_t ws_size,
                              hipStream_t stream) {
  const float* sx = (const float*)d_in[0];
  const float* sea = (const float*)d_in[1];
  const float* sg = (const float*)d_in[2];
  const float* vx = (const float*)d_in[3];
  const float* vea = (const float*)d_in[4];
  const float* vg = (const float*)d_in[5];
  const int* sei = (const int*)d_in[n_in - 4];
  const int* sbatch = (const int*)d_in[n_in - 3];
  const int* vei = (const int*)d_in[n_in - 2];
  const int* vbatch = (const int*)d_in[n_in - 1];
  const int n1 = in_sizes[0] / 64, e1 = in_sizes[1] / 16;
  const int n2 = in_sizes[3] / 64, e2 = in_sizes[4] / 16;
  const int nmax = n1 > n2 ? n1 : n2, emax = e1 > e2 ? e1 : e2;
  auto P = [&](int i) { return (const float*)d_in[i]; };

  // bump-allocate scratch (fp32)
  float* ws = (float*)d_ws;
  size_t off = 0;
  auto alloc = [&](size_t cnt) { float* p = ws + off; off += cnt; return p; };
  Scratch w;
  w.h     = alloc((size_t)nmax * HC);
  w.xw    = alloc((size_t)nmax * HC);
  w.convo = alloc((size_t)nmax * HC);
  w.R     = alloc((size_t)nmax * HC);
  w.wT    = alloc((size_t)CATD * 512);   // max weight: 552x512
  w.a_s   = alloc((size_t)nmax * 4);
  w.a_d   = alloc((size_t)nmax * 4);
  w.mmax  = alloc((size_t)nmax * 4);
  w.ssum  = alloc((size_t)nmax * 4);
  w.a_e   = alloc((size_t)emax * 4);
  w.alpha = alloc((size_t)emax * 4);
  w.Me    = alloc(64);
  w.xcat  = alloc((size_t)NG * CATD);
  w.fA    = alloc((size_t)NG * 512);
  w.fB    = alloc((size_t)NG * 512);
  w.fR    = alloc((size_t)NG * 512);
  w.fT    = alloc((size_t)NG * 512);
  w.stT   = alloc(2 * 512);
  w.stR   = alloc(2 * 512);
  (void)ws_size;

  // build concat input [512, 552]: [pool1 | sg | pool2 | vg]
  k_fill<<<cdiv((long)NG * CATD, 256), 256, 0, stream>>>(w.xcat, 0.f, (long)NG * CATD);
  k_copycols<<<cdiv(NG * GDIM, 256), 256, 0, stream>>>(sg, w.xcat, NG, GDIM, HC);
  k_copycols<<<cdiv(NG * GDIM, 256), 256, 0, stream>>>(vg, w.xcat, NG, GDIM, HC + GDIM + HC);

  run_backbone(d_in, 20, sx, sea, sei, sbatch, n1, e1, 0, w, stream);           // g1
  run_backbone(d_in, 64, vx, vea, vei, vbatch, n2, e2, HC + GDIM, w, stream);   // g2

  // FC head: x -> bn(relu(fc_i(x))) + bn(rfc_i(x)), i=1..3, then fc4
  const float* xin = w.xcat;
  int Kc = CATD;
  float* ping[2] = {w.fA, w.fB};
  const dim3 gFC(cdiv((long)NG * 512, 256));
  for (int i = 1; i <= 3; ++i) {
    int rfc = 114 + (i - 1) * 2, fcp = 12 + (i - 1) * 2;
    int rbn = 108 + (i - 1) * 2, fbn = 6 + (i - 1) * 2;
    gemmT(xin, P(rfc), P(rfc + 1), w.fR, NG, 512, Kc, w, stream);
    k_fill<<<dim3(4), 256, 0, stream>>>(w.stR, 0.f, 2 * 512);
    k_colstats<<<dim3(256), 256, 0, stream>>>(w.fR, w.stR, NG, 512);
    gemmT(xin, P(fcp), P(fcp + 1), w.fT, NG, 512, Kc, w, stream);
    k_bias_relu<<<gFC, 256, 0, stream>>>(w.fT, nullptr, (long)NG * 512, 512);
    k_fill<<<dim3(4), 256, 0, stream>>>(w.stT, 0.f, 2 * 512);
    k_colstats<<<dim3(256), 256, 0, stream>>>(w.fT, w.stT, NG, 512);
    float* xo = ping[(i - 1) & 1];
    k_bn2_add<<<gFC, 256, 0, stream>>>(w.fT, w.stT, P(fbn + 1), P(fbn),
                                       w.fR, w.stR, P(rbn + 1), P(rbn),
                                       xo, NG, 512);
    xin = xo;
    Kc = 512;
  }
  k_fc4<<<dim3(2), 256, 0, stream>>>(xin, P(18), P(19), (float*)d_out, NG, 512);
  (void)out_size;
}
